// HeterogeneousFamilyGNN_75093208203879
// MI455X (gfx1250) — matrix operand order repaired and checked by reading, not compiled, not run on hardware
//
#include <hip/hip_runtime.h>

// ---------------------------------------------------------------------------
// HeterogeneousFamilyGNN forward for MI455X (gfx1250, wave32, WMMA).
// GEMMs: v_wmma_f32_16x16x32_f16, B panel staged in LDS via
//        global_load_async_to_lds_b128 (ASYNCcnt) + ds_load_b128.
// Graph phase: atomic-based segment softmax + scatter-add.
// ---------------------------------------------------------------------------

#define N_NODES  100000
#define HIDDEN   256
#define EP_EDGES 200000
#define ES_EDGES 100000
#define SLOPE    0.2f

typedef __attribute__((ext_vector_type(16))) _Float16 v16h;
typedef __attribute__((ext_vector_type(8)))  float    v8f;

// ---------------- order-preserving float<->uint key for atomicMax ----------
__device__ __forceinline__ unsigned fenc(float f) {
    unsigned u = __float_as_uint(f);
    return (u & 0x80000000u) ? ~u : (u | 0x80000000u);
}
__device__ __forceinline__ float fdec(unsigned k) {
    unsigned u = (k & 0x80000000u) ? (k & 0x7FFFFFFFu) : ~k;
    return __uint_as_float(u);
}
__device__ __forceinline__ float leaky(float x) { return x > 0.0f ? x : x * SLOPE; }

// ---------------- gfx1250 async global->LDS copy (ASYNCcnt path) -----------
__device__ __forceinline__ void async_g2l_b128(unsigned lds_off, const void* gptr) {
    // LDS dest address VGPR first, 64-bit global address VGPR pair second.
    asm volatile("global_load_async_to_lds_b128 %0, %1, off"
                 :: "v"(lds_off), "v"(gptr) : "memory");
}
__device__ __forceinline__ void wait_asynccnt0() {
    asm volatile("s_wait_asynccnt 0x0" ::: "memory");
}

// ---------------- weight prep: W[k][n] f32 -> Wt[n][k] f16 -----------------
__global__ __launch_bounds__(256) void prep_wt(const float* __restrict__ W,
                                               _Float16* __restrict__ Wt) {
    int idx = blockIdx.x * 256 + threadIdx.x;     // 65536 elements
    int k = idx >> 8, n = idx & 255;
    Wt[(size_t)n * HIDDEN + k] = (_Float16)W[(size_t)k * HIDDEN + n];
}

// ---------------- WMMA GEMM: D[M,256] = act(A[M,256] @ Wt^T + bias) --------
// Block: 256 threads = 8 waves -> 128 rows x 64 cols.
// Wave: one 16-row x 64-col strip = 4 accumulators, 4 WMMA per K-step.
// B panel (64 cols x 256 K f16 = 32KB) staged in LDS with async loads.
__global__ __launch_bounds__(256) void gemm256_wmma(
    const float* __restrict__ A, const _Float16* __restrict__ Wt,
    const float* __restrict__ bias, float* __restrict__ D, int M, int act)
{
    __shared__ _Float16 sB[64 * HIDDEN];          // [panel col][k], 32 KB

    const int tid     = threadIdx.x;
    const int colBase = blockIdx.y * 64;

    // ---- cooperative async stage of B panel: 2048 x 16B chunks ----
    {
        unsigned sbase = (unsigned)(size_t)(__attribute__((address_space(3))) _Float16*)sB;
        for (int i = tid; i < 2048; i += 256) {
            int col   = i >> 5;                    // 32 chunks per column
            int koff  = (i & 31) * 8;              // halfs
            const void* g = (const void*)(Wt + (size_t)(colBase + col) * HIDDEN + koff);
            async_g2l_b128(sbase + (unsigned)(col * HIDDEN + koff) * 2u, g);
        }
    }
    wait_asynccnt0();
    __syncthreads();

    const int wave = tid >> 5;
    const int lid  = tid & 31;
    const int l    = lid & 15;          // position within half-wave
    const int g    = lid >> 4;          // half-wave group
    const int row0 = blockIdx.x * 128 + wave * 16;

    if (row0 < M) {
        const float* __restrict__ Arow = A + (size_t)(row0 + l) * HIDDEN;

        v8f acc0 = {}, acc1 = {}, acc2 = {}, acc3 = {};
        #pragma unroll
        for (int k0 = 0; k0 < HIDDEN; k0 += 32) {
            // A fragment (contiguous k order, same as B): lane l+16g holds
            //   elems 0..15 -> K = k0 + 16g .. +15
            float4 f0 = *(const float4*)(Arow + k0 + 16 * g);
            float4 f1 = *(const float4*)(Arow + k0 + 16 * g + 4);
            float4 f2 = *(const float4*)(Arow + k0 + 16 * g + 8);
            float4 f3 = *(const float4*)(Arow + k0 + 16 * g + 12);
            v16h a;
            a[0]=(_Float16)f0.x;  a[1]=(_Float16)f0.y;  a[2]=(_Float16)f0.z;  a[3]=(_Float16)f0.w;
            a[4]=(_Float16)f1.x;  a[5]=(_Float16)f1.y;  a[6]=(_Float16)f1.z;  a[7]=(_Float16)f1.w;
            a[8]=(_Float16)f2.x;  a[9]=(_Float16)f2.y;  a[10]=(_Float16)f2.z; a[11]=(_Float16)f2.w;
            a[12]=(_Float16)f3.x; a[13]=(_Float16)f3.y; a[14]=(_Float16)f3.z; a[15]=(_Float16)f3.w;
            // B fragments from LDS (32x16 f16): lane l+16g: col N=l, K=k0+16g..+15
            v16h b0 = *(const v16h*)&sB[(0 * 16 + l) * HIDDEN + k0 + 16 * g];
            v16h b1 = *(const v16h*)&sB[(1 * 16 + l) * HIDDEN + k0 + 16 * g];
            v16h b2 = *(const v16h*)&sB[(2 * 16 + l) * HIDDEN + k0 + 16 * g];
            v16h b3 = *(const v16h*)&sB[(3 * 16 + l) * HIDDEN + k0 + 16 * g];
            acc0 = __builtin_amdgcn_wmma_f32_16x16x32_f16(false, a, false, b0, (short)0, acc0, false, false);
            acc1 = __builtin_amdgcn_wmma_f32_16x16x32_f16(false, a, false, b1, (short)0, acc1, false, false);
            acc2 = __builtin_amdgcn_wmma_f32_16x16x32_f16(false, a, false, b2, (short)0, acc2, false, false);
            acc3 = __builtin_amdgcn_wmma_f32_16x16x32_f16(false, a, false, b3, (short)0, acc3, false, false);
        }

        // C/D layout: VGPR v, lane l+16g -> D[row0 + v + 8g][colBase + t*16 + l]
        #pragma unroll
        for (int t = 0; t < 4; ++t) {
            v8f acc = (t == 0) ? acc0 : (t == 1) ? acc1 : (t == 2) ? acc2 : acc3;
            const int col = colBase + t * 16 + l;
            float bv = bias ? bias[col] : 0.0f;
            #pragma unroll
            for (int v = 0; v < 8; ++v) {
                int m = row0 + v + 8 * g;
                float val = acc[v] + bv;
                if (act) val = fmaxf(val, 0.0f);
                D[(size_t)m * HIDDEN + col] = val;
            }
        }
    }
}

// ---------------- fills ----------------------------------------------------
__global__ __launch_bounds__(256) void fill_f32(float* p, float v, int n) {
    int i = blockIdx.x * 256 + threadIdx.x;
    if (i < n) p[i] = v;
}
__global__ __launch_bounds__(256) void fill_u32(unsigned* p, unsigned v, int n) {
    int i = blockIdx.x * 256 + threadIdx.x;
    if (i < n) p[i] = v;
}

// ---------------- per-node attention coefficients --------------------------
// aS[n,h] = sum_c hT[n,h,c]*a_src[h,c] ; aD likewise
__global__ __launch_bounds__(256) void attn_coeff(
    const float* __restrict__ hT, const float* __restrict__ asrc,
    const float* __restrict__ adst, float* __restrict__ aS, float* __restrict__ aD,
    int H, int C, int total)
{
    int idx = blockIdx.x * 256 + threadIdx.x;
    if (idx >= total) return;
    int node = idx / H, hh = idx - node * H;
    const float* v  = hT   + (size_t)node * HIDDEN + hh * C;
    const float* ps = asrc + hh * C;
    const float* pd = adst + hh * C;
    float s = 0.0f, d = 0.0f;
    for (int c = 0; c < C; ++c) { float x = v[c]; s = fmaf(x, ps[c], s); d = fmaf(x, pd[c], d); }
    aS[idx] = s; aD[idx] = d;
}

// ---------------- edge passes ----------------------------------------------
__global__ __launch_bounds__(256) void edge_max(
    const int* __restrict__ ei, const float* __restrict__ aS,
    const float* __restrict__ aD, unsigned* __restrict__ mxk, int E, int H)
{
    int idx = blockIdx.x * 256 + threadIdx.x;
    if (idx >= E * H) return;
    int e = idx / H, hh = idx - e * H;
    int s = ei[e], d = ei[E + e];
    float lg = leaky(aS[s * H + hh] + aD[d * H + hh]);
    atomicMax(&mxk[d * H + hh], fenc(lg));
}

__global__ __launch_bounds__(256) void seg_max_fix(unsigned* __restrict__ mxk, int n) {
    int i = blockIdx.x * 256 + threadIdx.x;
    if (i >= n) return;
    float m = fdec(mxk[i]);
    float inf = __builtin_inff();
    ((float*)mxk)[i] = (m > -inf && m < inf) ? m : 0.0f;   // where(isfinite, m, 0)
}

__global__ __launch_bounds__(256) void edge_den(
    const int* __restrict__ ei, const float* __restrict__ aS,
    const float* __restrict__ aD, const float* __restrict__ mx,
    float* __restrict__ den, int E, int H)
{
    int idx = blockIdx.x * 256 + threadIdx.x;
    if (idx >= E * H) return;
    int e = idx / H, hh = idx - e * H;
    int s = ei[e], d = ei[E + e];
    int di = d * H + hh;
    float lg = leaky(aS[s * H + hh] + aD[di]);
    atomicAdd(&den[di], expf(lg - mx[di]));
}

__global__ __launch_bounds__(256) void edge_msg(
    const int* __restrict__ ei, const float* __restrict__ aS,
    const float* __restrict__ aD, const float* __restrict__ mx,
    const float* __restrict__ den, const float* __restrict__ hT,
    float* __restrict__ agg, int E, int H, int C)
{
    int idx = blockIdx.x * 256 + threadIdx.x;
    if (idx >= E * H) return;
    int e = idx / H, hh = idx - e * H;
    int s = ei[e], d = ei[E + e];
    int di = d * H + hh;
    float lg = leaky(aS[s * H + hh] + aD[di]);
    float alpha = expf(lg - mx[di]) / (den[di] + 1e-16f);
    const float* hv  = hT  + (size_t)s * HIDDEN + hh * C;
    float*       out = agg + (size_t)d * HIDDEN + hh * C;
    for (int c = 0; c < C; ++c) atomicAdd(&out[c], hv[c] * alpha);
}

// ---------------- layer finalize: h = relu(agg + bP + bS) ------------------
__global__ __launch_bounds__(256) void finalize_layer(
    const float* __restrict__ agg, const float* __restrict__ bP,
    const float* __restrict__ bS, float* __restrict__ h, int total)
{
    int i = blockIdx.x * 256 + threadIdx.x;
    if (i >= total) return;
    int col = i & 255;
    h[i] = fmaxf(agg[i] + bP[col] + bS[col], 0.0f);
}

// ---------------------------------------------------------------------------
extern "C" void kernel_launch(void* const* d_in, const int* in_sizes, int n_in,
                              void* d_out, int out_size, void* d_ws, size_t ws_size,
                              hipStream_t stream)
{
    (void)in_sizes; (void)n_in; (void)out_size; (void)ws_size;
    // setup_inputs() insertion order:
    // 0:x 1:ei_parent 2:ei_spouse 3:emb_w 4:emb_b
    // 5..36: layers[i in 0..3][parent,spouse]{W,a_src,a_dst,b}
    // 37:father_w 38:father_b 39:mother_w 40:mother_b
    const float* x        = (const float*)d_in[0];
    const int*   eiP      = (const int*)d_in[1];
    const int*   eiS      = (const int*)d_in[2];
    const float* emb_w    = (const float*)d_in[3];
    const float* emb_b    = (const float*)d_in[4];
    const float* father_w = (const float*)d_in[37];
    const float* father_b = (const float*)d_in[38];
    const float* mother_w = (const float*)d_in[39];
    const float* mother_b = (const float*)d_in[40];

    const size_t NB = (size_t)N_NODES * HIDDEN;
    float* out = (float*)d_out;
    float* h   = out;            // final h slot; holds activations all along
    float* agg = out + NB;       // father slot; reused as aggregation buffer
    float* hT  = out + 2 * NB;   // mother slot; reused as per-relation features

    // workspace: aS, aD, mx(keys/float), den  each N*8 f32; then f16 weight slot
    float*    aS  = (float*)d_ws;
    float*    aD  = aS + (size_t)N_NODES * 8;
    unsigned* mxk = (unsigned*)(aD + (size_t)N_NODES * 8);
    float*    mx  = (float*)mxk;
    float*    den = (float*)(mxk + (size_t)N_NODES * 8);
    _Float16* Wt  = (_Float16*)(den + (size_t)N_NODES * 8);

    const unsigned NEG_INF_KEY = 0x007FFFFFu;   // fenc(-inf)
    const int NH_MAX_BLK = ((int)NB + 255) / 256;

    auto gemm = [&](const float* A, const float* W, const float* bias,
                    float* Dst, int act) {
        prep_wt<<<256, 256, 0, stream>>>(W, Wt);
        dim3 grid((N_NODES + 127) / 128, 4);    // 128 rows x 64 cols per block
        gemm256_wmma<<<grid, 256, 0, stream>>>(A, Wt, bias, Dst, N_NODES, act);
    };

    // node-type embedding: h = relu(x @ emb_w + emb_b)
    gemm(x, emb_w, emb_b, h, 1);

    for (int i = 0; i < 4; ++i) {
        const int H = (i < 3) ? 8 : 1;
        const int C = HIDDEN / H;
        fill_f32<<<NH_MAX_BLK, 256, 0, stream>>>(agg, 0.0f, (int)NB);
        for (int et = 0; et < 2; ++et) {
            const float* W    = (const float*)d_in[5 + i * 8 + et * 4 + 0];
            const float* asrc = (const float*)d_in[5 + i * 8 + et * 4 + 1];
            const float* adst = (const float*)d_in[5 + i * 8 + et * 4 + 2];
            const int*   ei   = (et == 0) ? eiP : eiS;
            const int    E    = (et == 0) ? EP_EDGES : ES_EDGES;
            const int    nh   = N_NODES * H;
            const int    eh   = E * H;

            gemm(h, W, nullptr, hT, 0);                                   // hT = h @ W
            attn_coeff<<<(nh + 255) / 256, 256, 0, stream>>>(hT, asrc, adst, aS, aD, H, C, nh);
            fill_u32<<<(nh + 255) / 256, 256, 0, stream>>>(mxk, NEG_INF_KEY, nh);
            fill_f32<<<(nh + 255) / 256, 256, 0, stream>>>(den, 0.0f, nh);
            edge_max<<<(eh + 255) / 256, 256, 0, stream>>>(ei, aS, aD, mxk, E, H);
            seg_max_fix<<<(nh + 255) / 256, 256, 0, stream>>>(mxk, nh);
            edge_den<<<(eh + 255) / 256, 256, 0, stream>>>(ei, aS, aD, mx, den, E, H);
            edge_msg<<<(eh + 255) / 256, 256, 0, stream>>>(ei, aS, aD, mx, den, hT, agg, E, H, C);
        }
        const float* bP = (const float*)d_in[5 + i * 8 + 3];
        const float* bS = (const float*)d_in[5 + i * 8 + 7];
        finalize_layer<<<NH_MAX_BLK, 256, 0, stream>>>(agg, bP, bS, h, (int)NB);
    }

    // output heads (agg/hT scratch roles are finished; these ARE their slots)
    gemm(h, father_w, father_b, agg, 0);   // father -> d_out[NB : 2NB]
    gemm(h, mother_w, mother_b, hT,  0);   // mother -> d_out[2NB : 3NB]
}
